// SAGE_JK_20504173871206
// MI455X (gfx1250) — matrix operand order repaired and checked by reading, not compiled, not run on hardware
//
#include <hip/hip_runtime.h>
#include <hip/hip_bf16.h>

typedef __attribute__((ext_vector_type(16))) _Float16 v16h;
typedef __attribute__((ext_vector_type(8)))  _Float16 v8h;
typedef __attribute__((ext_vector_type(8)))  float    v8f;

#define Nn 50000
#define Ee 800000
#define Gg 64
#define Cc 64
#define JK 192
#define NE (Ee + Nn)
#define EPS_BN 1e-5f

// ---------- helpers ----------
__device__ __forceinline__ unsigned f2o(float f) {
    unsigned u = __float_as_uint(f);
    return (u & 0x80000000u) ? ~u : (u | 0x80000000u);
}
__device__ __forceinline__ float o2f(unsigned u) {
    return (u & 0x80000000u) ? __uint_as_float(u & 0x7fffffffu)
                             : __uint_as_float(~u);
}

// ---------- pack weights into WMMA B-fragment order (f32 -> f16) ----------
// Packed layout: Bp[(ks*NW + wave)*512 + lane*16 + i] = W[K*ncols + col]
// with K = ks*32 + 8*(lane>>4) + (i&7) + 16*(i>>3), col = wave*16 + (lane&15)   (ISA 7.12.2)
__global__ void repack_b_kernel(const float* __restrict__ W,
                                _Float16* __restrict__ Bp,
                                int kdim, int ncols) {
    int t = blockIdx.x * blockDim.x + threadIdx.x;
    if (t >= kdim * ncols) return;
    int nwaves = ncols >> 4;
    int tile = t >> 9;          // (ks, wave) tile of 32 lanes x 16 halfs
    int rem  = t & 511;
    int lane = rem >> 4;
    int i    = rem & 15;
    int ks   = tile / nwaves;
    int wv   = tile - ks * nwaves;
    int K    = ks * 32 + 8 * (lane >> 4) + (i & 7) + 16 * (i >> 3);
    int col  = wv * 16 + (lane & 15);
    Bp[t] = (_Float16)W[(size_t)K * ncols + col];
}

// ---------- WMMA GEMM: C[MxNCOLS] = relu?(A[MxKDIM] @ B + bias) ----------
// grid.x = M/16 row tiles; blockDim.x = 32*(NCOLS/16); wave w owns cols
// [16w,16w+16). A tile staged in LDS as f16; B pre-packed in fragment order.
template <int KDIM, int NCOLS>
__global__ void wmma_gemm_kernel(const float* __restrict__ A, int lda,
                                 const _Float16* __restrict__ Bp,
                                 const float* __restrict__ bias, int relu,
                                 float* __restrict__ Cout, int ldc) {
    constexpr int NW = NCOLS / 16;
    constexpr int LSTR = KDIM + 8;          // pad: keeps 16B alignment, spreads banks
    __shared__ _Float16 As[16 * LSTR];
    const int rowBase = blockIdx.x * 16;

    // cooperative stage of the 16 x KDIM A tile (f32 -> f16), float4 granules
    for (int idx = threadIdx.x; idx < 16 * KDIM / 4; idx += 32 * NW) {
        int row = idx / (KDIM / 4);
        int c4  = idx - row * (KDIM / 4);
        float4 v = *(const float4*)(A + (size_t)(rowBase + row) * lda + c4 * 4);
        _Float16* dst = &As[row * LSTR + c4 * 4];
        dst[0] = (_Float16)v.x; dst[1] = (_Float16)v.y;
        dst[2] = (_Float16)v.z; dst[3] = (_Float16)v.w;
    }
    __syncthreads();

    const int wave = threadIdx.x >> 5;
    const int lane = threadIdx.x & 31;
    const int g = lane >> 4, lr = lane & 15;

    v8f acc = {};
#pragma unroll
    for (int ks = 0; ks < KDIM / 32; ++ks) {
        // A fragment: row lr, K runs [32ks+8g .. +7] and [32ks+16+8g .. +7]
        const _Float16* arow = &As[lr * LSTR + ks * 32 + 8 * g];
        v8h alo = *(const v8h*)arow;
        v8h ahi = *(const v8h*)(arow + 16);
        v16h a = __builtin_shufflevector(alo, ahi, 0, 1, 2, 3, 4, 5, 6, 7,
                                         8, 9, 10, 11, 12, 13, 14, 15);
        // B fragment: contiguous 16 halfs per lane in packed buffer
        const _Float16* bptr = Bp + ((size_t)ks * NW + wave) * 512 + lane * 16;
        v8h blo = *(const v8h*)bptr;
        v8h bhi = *(const v8h*)(bptr + 8);
        v16h b = __builtin_shufflevector(blo, bhi, 0, 1, 2, 3, 4, 5, 6, 7,
                                         8, 9, 10, 11, 12, 13, 14, 15);
        acc = __builtin_amdgcn_wmma_f32_16x16x32_f16(
            false, a, false, b, (short)0, acc, false, false);
    }
    // D layout: VGPR r holds (row = r + 8*g, col = lane%16)
    const int col = wave * 16 + lr;
    const float bv = bias ? bias[col] : 0.0f;
#pragma unroll
    for (int r = 0; r < 8; ++r) {
        int row = rowBase + r + 8 * g;
        float v = acc[r] + bv;
        if (relu) v = fmaxf(v, 0.0f);
        Cout[(size_t)row * ldc + col] = v;
    }
}

// ---------- per-node attention scalars ----------
__global__ void att_kernel(const float* __restrict__ h,
                           const float* __restrict__ att_src,
                           const float* __restrict__ att_dst,
                           float* __restrict__ a_s, float* __restrict__ a_d) {
    int i = blockIdx.x * blockDim.x + threadIdx.x;
    if (i >= Nn) return;
    const float4* hp = (const float4*)(h + (size_t)i * Cc);
    float s = 0.0f, d = 0.0f;
#pragma unroll
    for (int k = 0; k < Cc / 4; ++k) {
        float4 hv = hp[k];
        float4 as = ((const float4*)att_src)[k];
        float4 ad = ((const float4*)att_dst)[k];
        s += hv.x * as.x + hv.y * as.y + hv.z * as.z + hv.w * as.w;
        d += hv.x * ad.x + hv.y * ad.y + hv.z * ad.z + hv.w * ad.w;
    }
    a_s[i] = s; a_d[i] = d;
}

// ---------- per-layer scratch init ----------
__global__ void init_layer_kernel(float* __restrict__ agg,
                                  unsigned* __restrict__ m_u32,
                                  float* __restrict__ denom) {
    int t = blockIdx.x * blockDim.x + threadIdx.x;
    if (t < Nn * Cc) agg[t] = 0.0f;
    if (t < Nn) { m_u32[t] = 0u; denom[t] = 0.0f; }
}

// ---------- edge pass 1: e = leaky_relu, segment max over dst ----------
__global__ void edge_max_kernel(const int* __restrict__ ei,
                                const float* __restrict__ a_s,
                                const float* __restrict__ a_d,
                                float* __restrict__ e_buf,
                                unsigned* __restrict__ m_u32) {
    int t = blockIdx.x * blockDim.x + threadIdx.x;
    if (t >= NE) return;
    int s, d;
    if (t < Ee) { s = ei[t]; d = ei[Ee + t]; } else { s = d = t - Ee; }
    float e = a_s[s] + a_d[d];
    e = e > 0.0f ? e : 0.2f * e;
    e_buf[t] = e;
    atomicMax(&m_u32[d], f2o(e));
}

// ---------- edge pass 2: ex = exp(e-m); denom += ex; agg[dst] += ex*h[src] ----------
// one wave32 per edge; each lane handles 2 channels (float2)
__global__ void edge_scatter_kernel(const int* __restrict__ ei,
                                    const float* __restrict__ e_buf,
                                    const unsigned* __restrict__ m_u32,
                                    const float* __restrict__ h,
                                    float* __restrict__ denom,
                                    float* __restrict__ agg) {
    int t = blockIdx.x * blockDim.x + threadIdx.x;
    int eid = t >> 5;
    int lane = t & 31;
    if (eid >= NE) return;
    int s, d;
    if (eid < Ee) { s = ei[eid]; d = ei[Ee + eid]; } else { s = d = eid - Ee; }
    float m  = o2f(m_u32[d]);
    float ex = __expf(e_buf[eid] - m);
    if (lane == 0) atomicAdd(&denom[d], ex);
    float2 hv = ((const float2*)(h + (size_t)s * Cc))[lane];
    atomicAdd(&agg[(size_t)d * Cc + 2 * lane],     hv.x * ex);
    atomicAdd(&agg[(size_t)d * Cc + 2 * lane + 1], hv.y * ex);
}

// ---------- per-node finalize: softmax denom, bias, BN(eval), ReLU ----------
__global__ void node_bn_kernel(const float* __restrict__ agg,
                               const float* __restrict__ denom,
                               const float* __restrict__ bias,
                               const float* __restrict__ gamma,
                               const float* __restrict__ beta,
                               const float* __restrict__ rm,
                               const float* __restrict__ rv,
                               float* __restrict__ xjk_slice) {
    int t = blockIdx.x * blockDim.x + threadIdx.x;
    if (t >= Nn * Cc) return;
    int node = t >> 6, c = t & 63;
    float v = agg[t] / denom[node] + bias[c];
    v = gamma[c] * (v - rm[c]) * rsqrtf(rv[c] + EPS_BN) + beta[c];
    v = fmaxf(v, 0.0f);
    xjk_slice[(size_t)node * JK + c] = v;
}

// ---------- lin3 + delay ----------
__global__ void lin3_kernel(const float* __restrict__ t2,
                            const float* __restrict__ W3,
                            const float* __restrict__ b3,
                            const float* __restrict__ x,
                            float* __restrict__ node_out,
                            float* __restrict__ node_delay) {
    int i = blockIdx.x * blockDim.x + threadIdx.x;
    if (i >= Nn) return;
    const float4* tp = (const float4*)(t2 + (size_t)i * 32);
    float acc = b3[0];
#pragma unroll
    for (int k = 0; k < 8; ++k) {
        float4 v = tp[k];
        float4 w = ((const float4*)W3)[k];
        acc += v.x * w.x + v.y * w.y + v.z * w.z + v.w * w.w;
    }
    node_out[i]   = acc;
    node_delay[i] = acc * x[(size_t)i * 64];   // delay_param = x[:,0]
}

// ---------- pooling ----------
__global__ void pool_init_kernel(float* __restrict__ sum_pool,
                                 float* __restrict__ sum_delay,
                                 float* __restrict__ cnt) {
    int t = blockIdx.x * blockDim.x + threadIdx.x;
    if (t < Gg * JK) sum_pool[t] = 0.0f;
    if (t < Gg) { sum_delay[t] = 0.0f; cnt[t] = 0.0f; }
}

__global__ void pool_kernel(const float* __restrict__ xjk,
                            const float* __restrict__ node_delay,
                            const int* __restrict__ batch,
                            float* __restrict__ sum_pool,
                            float* __restrict__ sum_delay,
                            float* __restrict__ cnt) {
    int t = blockIdx.x * blockDim.x + threadIdx.x;
    if (t >= Nn * JK) return;
    int node = t / JK, c = t - node * JK;
    int b = batch[node];
    atomicAdd(&sum_pool[b * JK + c], xjk[t]);
    if (c == 0) {
        atomicAdd(&sum_delay[b], node_delay[node]);
        atomicAdd(&cnt[b], 1.0f);
    }
}

// ---------- regression head: one block (32 lanes) per graph ----------
__global__ void reg_kernel(const float* __restrict__ other,
                           const float* __restrict__ sum_pool,
                           const float* __restrict__ sum_delay,
                           const float* __restrict__ cnt,
                           const float* __restrict__ W1,
                           const float* __restrict__ b1,
                           const float* __restrict__ W2,
                           const float* __restrict__ b2,
                           float* __restrict__ out_reg,
                           float* __restrict__ out_last) {
    int g = blockIdx.x;       // graph
    int j = threadIdx.x;      // 0..31 hidden unit
    float c = fmaxf(cnt[g], 1.0f);
    float acc = b1[j];
    for (int k = 0; k < 17; ++k) acc += other[g * 18 + k] * W1[k * 32 + j];
    acc += (sum_delay[g] / c) * W1[17 * 32 + j];
    for (int k = 0; k < JK; ++k)
        acc += (sum_pool[g * JK + k] / c) * W1[(18 + k) * 32 + j];
    acc = fmaxf(acc, 0.0f);
    float r = acc * W2[j];
    for (int off = 16; off > 0; off >>= 1) r += __shfl_down(r, off);
    if (j == 0) {
        out_reg[g]  = r + b2[0];
        out_last[g] = other[g * 18 + 17];
    }
}

static inline int cdiv(long long a, long long b) { return (int)((a + b - 1) / b); }

extern "C" void kernel_launch(void* const* d_in, const int* in_sizes, int n_in,
                              void* d_out, int out_size, void* d_ws, size_t ws_size,
                              hipStream_t stream) {
    (void)in_sizes; (void)n_in; (void)out_size; (void)ws_size;
    // --- inputs (setup_inputs order; params flattened as jax pytree: sorted keys) ---
    const float* x     = (const float*)d_in[0];
    const int*   ei    = (const int*)  d_in[1];
    const int*   batch = (const int*)  d_in[2];
    const float* other = (const float*)d_in[3];
    int p = 4;
    const float *bn_beta[3], *bn_gamma[3], *bn_rm[3], *bn_rv[3];
    for (int l = 0; l < 3; ++l) {          // 'bns': beta, gamma, rm, rv
        bn_beta[l]  = (const float*)d_in[p++];
        bn_gamma[l] = (const float*)d_in[p++];
        bn_rm[l]    = (const float*)d_in[p++];
        bn_rv[l]    = (const float*)d_in[p++];
    }
    const float *cW[3], *c_ad[3], *c_as[3], *c_b[3];
    for (int l = 0; l < 3; ++l) {          // 'convs': W, att_dst, att_src, bias
        cW[l]   = (const float*)d_in[p++];
        c_ad[l] = (const float*)d_in[p++];
        c_as[l] = (const float*)d_in[p++];
        c_b[l]  = (const float*)d_in[p++];
    }
    const float* lin1W = (const float*)d_in[p++];
    const float* lin1b = (const float*)d_in[p++];
    const float* lin2W = (const float*)d_in[p++];
    const float* lin2b = (const float*)d_in[p++];
    const float* lin3W = (const float*)d_in[p++];
    const float* lin3b = (const float*)d_in[p++];
    const float* reg1W = (const float*)d_in[p++];
    const float* reg1b = (const float*)d_in[p++];
    const float* reg2W = (const float*)d_in[p++];
    const float* reg2b = (const float*)d_in[p++];

    // --- workspace layout (floats) ---
    float* ws = (float*)d_ws;
    size_t off = 0;
    float*    xjk        = ws + off; off += (size_t)Nn * JK;
    float*    hbuf       = ws + off; off += (size_t)Nn * Cc;
    float*    agg        = ws + off; off += (size_t)Nn * Cc;
    float*    a_s        = ws + off; off += Nn;
    float*    a_d        = ws + off; off += Nn;
    unsigned* m_u32      = (unsigned*)(ws + off); off += Nn;
    float*    denom      = ws + off; off += Nn;
    float*    e_buf      = ws + off; off += NE;
    _Float16* wf16       = (_Float16*)(ws + off); off += (JK * 128) / 2 + 64;
    float*    t1         = ws + off; off += (size_t)Nn * 128;
    float*    t2         = ws + off; off += (size_t)Nn * 32;
    float*    node_delay = ws + off; off += Nn;
    float*    sum_pool   = ws + off; off += (size_t)Gg * JK;
    float*    sum_delay  = ws + off; off += Gg;
    float*    cnt        = ws + off; off += Gg;

    float* out_node = (float*)d_out;          // [N]
    float* out_reg  = out_node + Nn;          // [G]
    float* out_last = out_reg + Gg;           // [G]

    const int B = 256;
    // ===== 3 GAT layers =====
    for (int l = 0; l < 3; ++l) {
        const float* A = (l == 0) ? x : (xjk + (size_t)(l - 1) * Cc);
        int lda = (l == 0) ? Cc : JK;
        repack_b_kernel<<<cdiv(Cc * Cc, B), B, 0, stream>>>(cW[l], wf16, Cc, Cc);
        wmma_gemm_kernel<Cc, Cc><<<Nn / 16, 128, 0, stream>>>(A, lda, wf16,
                                                              nullptr, 0, hbuf, Cc);
        att_kernel<<<cdiv(Nn, B), B, 0, stream>>>(hbuf, c_as[l], c_ad[l], a_s, a_d);
        init_layer_kernel<<<cdiv((long long)Nn * Cc, B), B, 0, stream>>>(agg, m_u32, denom);
        edge_max_kernel<<<cdiv(NE, B), B, 0, stream>>>(ei, a_s, a_d, e_buf, m_u32);
        edge_scatter_kernel<<<cdiv((long long)NE * 32, B), B, 0, stream>>>(
            ei, e_buf, m_u32, hbuf, denom, agg);
        node_bn_kernel<<<cdiv((long long)Nn * Cc, B), B, 0, stream>>>(
            agg, denom, c_b[l], bn_gamma[l], bn_beta[l], bn_rm[l], bn_rv[l],
            xjk + (size_t)l * Cc);
    }

    // ===== node MLP =====
    repack_b_kernel<<<cdiv(JK * 128, B), B, 0, stream>>>(lin1W, wf16, JK, 128);
    wmma_gemm_kernel<JK, 128><<<Nn / 16, 256, 0, stream>>>(xjk, JK, wf16,
                                                           lin1b, 1, t1, 128);
    repack_b_kernel<<<cdiv(128 * 32, B), B, 0, stream>>>(lin2W, wf16, 128, 32);
    wmma_gemm_kernel<128, 32><<<Nn / 16, 64, 0, stream>>>(t1, 128, wf16,
                                                          lin2b, 1, t2, 32);
    lin3_kernel<<<cdiv(Nn, B), B, 0, stream>>>(t2, lin3W, lin3b, x,
                                               out_node, node_delay);

    // ===== pooling + regression head =====
    pool_init_kernel<<<cdiv(Gg * JK, B), B, 0, stream>>>(sum_pool, sum_delay, cnt);
    pool_kernel<<<cdiv((long long)Nn * JK, B), B, 0, stream>>>(
        xjk, node_delay, batch, sum_pool, sum_delay, cnt);
    reg_kernel<<<Gg, 32, 0, stream>>>(other, sum_pool, sum_delay, cnt,
                                      reg1W, reg1b, reg2W, reg2b,
                                      out_reg, out_last);
}
